// MultiHeadAttentionBlock_54193897341142
// MI455X (gfx1250) — matrix operand order repaired and checked by reading, not compiled, not run on hardware
//
#include <hip/hip_runtime.h>
#include <hip/hip_bf16.h>

typedef __attribute__((ext_vector_type(16))) __bf16 bf16x16;
typedef __attribute__((ext_vector_type(8)))  float  f32x8;

#define HB 2
#define HS 2048
#define HD 1024
#define HH 16
#define HDK 64
#define HM (HB * HS)   // 4096

// ---------------- CDNA5 async copy helpers (§08_async_tensor) ----------------
// GLOBAL_LOAD_ASYNC_TO_LDS_B128: per-lane 16B global->LDS, tracked with ASYNCcnt.
// ASYNCcnt completes in order -> ping-pong buffers with a nonzero wait threshold.

static __device__ __forceinline__ void async_load_b128(unsigned lds_off, const void* gptr) {
  unsigned long long ga = (unsigned long long)(uintptr_t)gptr;
  asm volatile("global_load_async_to_lds_b128 %0, %1, off"
               :: "v"(lds_off), "v"(ga) : "memory");
}
static __device__ __forceinline__ void wait_async0()   { asm volatile("s_wait_asynccnt 0" ::: "memory"); }
static __device__ __forceinline__ void wait_async_le2(){ asm volatile("s_wait_asynccnt 2" ::: "memory"); }
static __device__ __forceinline__ void wait_async_le4(){ asm volatile("s_wait_asynccnt 4" ::: "memory"); }
static __device__ __forceinline__ void wait_ds0()      { asm volatile("s_wait_dscnt 0"    ::: "memory"); }
static __device__ __forceinline__ unsigned lds_off_of(const void* p) {
  return (unsigned)(uintptr_t)p;   // LDS aperture: addr[31:0] is the LDS offset
}

// ---------------- WMMA fragment helpers (CDNA5 §7.12.2 layouts) ----------------

// A operand: 16x32 (M x K) tile from row-major memory (row stride ldk elements).
static __device__ __forceinline__ bf16x16 load_fragA(const __bf16* base, int ldk, int koff) {
  const int lane = threadIdx.x & 31;
  const int row = lane & 15;
  const int half = lane >> 4;
  const __bf16* r = base + (size_t)row * ldk + koff;
  bf16x16 f;
#pragma unroll
  for (int v = 0; v < 8; ++v) {
    const int kb = half * 16 + 2 * v;
    f[2 * v]     = r[kb];
    f[2 * v + 1] = r[kb + 1];
  }
  return f;
}

// B operand: 32x16 (K x N), source row-major N x K (B transposed). Lane n holds
// column n; half-wave 0 K=0..15 sequential, half 1 K=16..31 (contiguous -> b128 loads).
static __device__ __forceinline__ bf16x16 load_fragB(const __bf16* base, int ldk, int koff) {
  const int lane = threadIdx.x & 31;
  const int col = lane & 15;
  const int half = lane >> 4;
  const __bf16* r = base + (size_t)col * ldk + koff;
  bf16x16 f;
#pragma unroll
  for (int v = 0; v < 8; ++v) {
    const int k = half * 16 + 2 * v;
    f[2 * v]     = r[k];
    f[2 * v + 1] = r[k + 1];
  }
  return f;
}

static __device__ __forceinline__ f32x8 wmma_bf16(bf16x16 a, bf16x16 b, f32x8 c) {
  return __builtin_amdgcn_wmma_f32_16x16x32_bf16(false, a, false, b, (short)0, c, false, false);
}

// ---------------- 0) f32 -> bf16 conversion ----------------

__global__ __launch_bounds__(256) void cvt_f32_bf16(const float* __restrict__ in,
                                                    __bf16* __restrict__ out, int n4) {
  int i = blockIdx.x * 256 + threadIdx.x;
  if (i < n4) {
    float4 v = ((const float4*)in)[i];
    __bf16* o = out + (size_t)i * 4;
    o[0] = (__bf16)v.x; o[1] = (__bf16)v.y; o[2] = (__bf16)v.z; o[3] = (__bf16)v.w;
  }
}

// ---------------- 1) GEMM: C(MxN,f32) = A(MxK,bf16) @ Bw(NxK,bf16)^T ----------------
// 256 threads = 8 waves; block tile 128x128; wave tile 32(M)x64(N) = 2x4 wmma tiles.
// Double-buffered LDS; K-slabs staged with GLOBAL_LOAD_ASYNC_TO_LDS_B128 and drained
// with s_wait_asynccnt 4 so the next slab's DMA overlaps the current slab's WMMAs.

__global__ __launch_bounds__(256) void gemm_bf16_nt(const __bf16* __restrict__ A,
                                                    const __bf16* __restrict__ Bw,
                                                    float* __restrict__ C,
                                                    int Mm, int Nn, int Kk) {
  __shared__ __attribute__((aligned(16))) __bf16 As[2][128 * 32];
  __shared__ __attribute__((aligned(16))) __bf16 Bs[2][128 * 32];

  const int t = threadIdx.x;
  const int wave = t >> 5;
  const int wm = wave >> 1;                 // 0..3 -> 32-row strip
  const int wn = wave & 1;                  // 0..1 -> 64-col strip
  const int m0 = blockIdx.y * 128;
  const int n0 = blockIdx.x * 128;
  const int lane = t & 31;
  const int col = lane & 15, half = lane >> 4;

  const unsigned asb = lds_off_of(As);
  const unsigned bsb = lds_off_of(Bs);

  // stage one 32-wide K slab into buffer `buf` (4 asyncs per thread)
  auto stage = [&](int buf, int kk) {
#pragma unroll
    for (int i = 0; i < 2; ++i) {
      const int idx = i * 256 + t;
      const int row = idx >> 2, c4 = idx & 3;          // 4 x 16B per 32-elem row
      async_load_b128(asb + buf * 8192 + idx * 16, A  + (size_t)(m0 + row) * Kk + kk + c4 * 8);
      async_load_b128(bsb + buf * 8192 + idx * 16, Bw + (size_t)(n0 + row) * Kk + kk + c4 * 8);
    }
  };

  const f32x8 fz = {0.f, 0.f, 0.f, 0.f, 0.f, 0.f, 0.f, 0.f};
  f32x8 acc[2][4];
#pragma unroll
  for (int i = 0; i < 2; ++i)
#pragma unroll
    for (int j = 0; j < 4; ++j) acc[i][j] = fz;

  const int nk = Kk >> 5;
  stage(0, 0);
  for (int ik = 0; ik < nk; ++ik) {
    const int cur = ik & 1;
    if (ik + 1 < nk) {
      stage(cur ^ 1, (ik + 1) * 32);      // issue next slab before draining current
      wait_async_le4();                   // release current slab only (in-order cnt)
      if (ik + 2 < nk) {                  // warm L2 two slabs ahead
        __builtin_prefetch(A  + (size_t)(m0 + (t >> 1)) * Kk + (ik + 2) * 32 + (t & 1) * 16, 0, 1);
        __builtin_prefetch(Bw + (size_t)(n0 + (t >> 1)) * Kk + (ik + 2) * 32 + (t & 1) * 16, 0, 1);
      }
    } else {
      wait_async0();
    }
    __syncthreads();

    const __bf16* Ab = As[cur];
    const __bf16* Bb = Bs[cur];
    bf16x16 a0 = load_fragA(Ab + (size_t)(wm * 32) * 32, 32, 0);
    bf16x16 a1 = load_fragA(Ab + (size_t)(wm * 32 + 16) * 32, 32, 0);
    bf16x16 b[4];
#pragma unroll
    for (int j = 0; j < 4; ++j)
      b[j] = load_fragB(Bb + (size_t)(wn * 64 + j * 16) * 32, 32, 0);
#pragma unroll
    for (int j = 0; j < 4; ++j) {
      acc[0][j] = wmma_bf16(a0, b[j], acc[0][j]);
      acc[1][j] = wmma_bf16(a1, b[j], acc[1][j]);
    }
    __syncthreads();                      // protect buffer before it is restaged
  }

#pragma unroll
  for (int i = 0; i < 2; ++i)
#pragma unroll
    for (int j = 0; j < 4; ++j) {
      const int cb = n0 + wn * 64 + j * 16 + col;
      const int rb = m0 + wm * 32 + i * 16 + half * 8;
#pragma unroll
      for (int r = 0; r < 8; ++r)
        C[(size_t)(rb + r) * Nn + cb] = acc[i][j][r];
    }
}

// ---------------- 2) per-head RMSNorm + RoPE + relayout to [B,H,S,DK] bf16 ----------

__global__ __launch_bounds__(256) void head_norm_rope(const float* __restrict__ X,
                                                      const float* __restrict__ nw,
                                                      __bf16* __restrict__ Y,
                                                      int do_norm_rope) {
  const int wid = blockIdx.x * 8 + (threadIdx.x >> 5);
  const int lane = threadIdx.x & 31;
  const int m = wid >> 4;          // 0..4095 = b*S + s
  const int h = wid & 15;
  const int b = m >> 11;
  const int s = m & 2047;

  const float* xr = X + (size_t)m * HD + h * HDK;
  float x0 = xr[lane], x1 = xr[lane + 32];
  float y0 = x0, y1 = x1;
  if (do_norm_rope) {
    float ss = x0 * x0 + x1 * x1;
#pragma unroll
    for (int off = 16; off > 0; off >>= 1) ss += __shfl_xor(ss, off, 32);
    const float r = rsqrtf(ss * (1.0f / 64.0f) + 1e-10f);
    const float w0 = nw ? nw[lane] : 1.0f;
    const float w1 = nw ? nw[lane + 32] : 1.0f;
    y0 = x0 * r * w0;
    y1 = x1 * r * w1;
    const float invf = __expf(-13.122363377f * ((float)lane * (1.0f / 32.0f)));
    const float ang = (float)s * invf;
    float sn, cs;
    __sincosf(ang, &sn, &cs);
    const float o0 = y0 * cs - y1 * sn;
    const float o1 = y1 * cs + y0 * sn;
    y0 = o0; y1 = o1;
  }
  __bf16* yr = Y + ((size_t)((b * HH + h) * HS) + s) * HDK;
  yr[lane]      = (__bf16)y0;
  yr[lane + 32] = (__bf16)y1;
}

// ---------------- 3) causal flash attention, all-WMMA, double-buffered staging ------
// 128 threads = 4 waves; wave owns 16 query rows (block covers 64). Block-uniform
// key-block loop (fully-masked tiles are numerically inert), so K/V tiles are staged
// cooperatively: K via async-to-LDS (ping-pong), V transposed into LDS for contiguous
// B-fragment reads.

__global__ __launch_bounds__(128) void flash_attn(const __bf16* __restrict__ Q,
                                                  const __bf16* __restrict__ Kt,
                                                  const __bf16* __restrict__ V,
                                                  __bf16* __restrict__ O) {
  __shared__ __attribute__((aligned(16))) __bf16 Ks[2][32 * 64];   // [key][d]
  __shared__ __attribute__((aligned(16))) __bf16 Vt[2][64 * 32];   // [d][key]
  __shared__ __attribute__((aligned(16))) float  sbuf[4][16 * 32];
  __shared__ __attribute__((aligned(16))) __bf16 pbuf[4][16 * 32];
  __shared__ float abuf[4][16];

  const int t = threadIdx.x;
  const int w = t >> 5;
  const int lane = t & 31;
  const int col = lane & 15, half = lane >> 4;
  const int bh = blockIdx.y;                       // b*16 + h
  const int qt = blockIdx.x;
  const int qbase = qt * 64 + w * 16;
  const size_t base = (size_t)bh * HS * HDK;
  const __bf16* Qb = Q + base;
  const __bf16* Kb = Kt + base;
  const __bf16* Vb = V + base;
  const unsigned ksb = lds_off_of(Ks);

  // stage one 32-key block into buffer `buf` (2 asyncs/thread for K, transpose for V)
  auto stage = [&](int buf, int kb) {
    const int k0 = kb * 32;
#pragma unroll
    for (int i = 0; i < 2; ++i) {
      const int idx = i * 128 + t;        // 256 x 16B
      const int krow = idx >> 3, c8 = idx & 7;
      async_load_b128(ksb + buf * 4096 + idx * 16, Kb + (size_t)(k0 + krow) * HDK + c8 * 8);
      uint4 pv = *(const uint4*)(Vb + (size_t)(k0 + krow) * HDK + c8 * 8);
      const __bf16* pe = (const __bf16*)&pv;
#pragma unroll
      for (int e = 0; e < 8; ++e) Vt[buf][(c8 * 8 + e) * 32 + krow] = pe[e];
    }
  };

  const bf16x16 aq0 = load_fragA(Qb + (size_t)qbase * HDK, HDK, 0);
  const bf16x16 aq1 = load_fragA(Qb + (size_t)qbase * HDK, HDK, 32);

  const f32x8 fz = {0.f, 0.f, 0.f, 0.f, 0.f, 0.f, 0.f, 0.f};
  f32x8 o[4] = {fz, fz, fz, fz};
  float mrun = -3.0e38f, lrun = 0.0f;

  const int row = col;                  // softmax row handled by this lane (dup per half)
  const int qrow = qbase + row;
  const int kbMax = 2 * qt + 1;         // block-uniform causal frontier

  float*  sw = sbuf[w];
  __bf16* pw = pbuf[w];
  float*  aw = abuf[w];

  stage(0, 0);
  for (int kb = 0; kb <= kbMax; ++kb) {
    const int k0 = kb * 32;
    const int cur = kb & 1;
    if (kb < kbMax) {
      stage(cur ^ 1, kb + 1);           // overlap next block's DMA with this block's math
      wait_async_le2();
    } else {
      wait_async0();
    }
    __syncthreads();

    // ---- scores: two 16x16 tiles, DK=64 in two wmma steps ----
    const __bf16* Kc = Ks[cur];
    const __bf16* Vc = Vt[cur];
#pragma unroll
    for (int tn = 0; tn < 2; ++tn) {
      bf16x16 bk0 = load_fragB(Kc + (size_t)(tn * 16) * 64, 64, 0);
      bf16x16 bk1 = load_fragB(Kc + (size_t)(tn * 16) * 64, 64, 32);
      f32x8 sc = fz;
      sc = wmma_bf16(aq0, bk0, sc);
      sc = wmma_bf16(aq1, bk1, sc);
#pragma unroll
      for (int r = 0; r < 8; ++r)
        sw[(half * 8 + r) * 32 + tn * 16 + col] = sc[r] * 0.125f;   // 1/sqrt(DK)
    }
    wait_ds0();

    // ---- online softmax row pass (each half-wave redundantly covers rows 0..15) ----
    float sv[32];
    float rmax = -3.0e38f;
#pragma unroll
    for (int j = 0; j < 32; ++j) {
      float x = sw[row * 32 + j];
      if (k0 + j > qrow) x = -3.0e38f;          // causal mask
      sv[j] = x;
      rmax = fmaxf(rmax, x);
    }
    const float mnew = fmaxf(mrun, rmax);
    const float alpha = __expf(mrun - mnew);
    float rsum = 0.0f;
#pragma unroll
    for (int j = 0; j < 32; ++j) {
      const float p = (k0 + j > qrow) ? 0.0f : __expf(sv[j] - mnew);
      rsum += p;
      pw[row * 32 + j] = (__bf16)p;
    }
    lrun = lrun * alpha + rsum;
    mrun = mnew;
    aw[row] = alpha;
    wait_ds0();

    // ---- rescale output accumulators by alpha(row) ----
#pragma unroll
    for (int r = 0; r < 8; ++r) {
      const float ar = aw[half * 8 + r];
#pragma unroll
      for (int dt = 0; dt < 4; ++dt) o[dt][r] *= ar;
    }

    // ---- P @ V : one A frag (16x32 probs) x four 16-wide d tiles from Vt ----
    const bf16x16 ap = load_fragA(pw, 32, 0);
#pragma unroll
    for (int dt = 0; dt < 4; ++dt) {
      bf16x16 bv = load_fragB(Vc + (size_t)(dt * 16) * 32, 32, 0);
      o[dt] = wmma_bf16(ap, bv, o[dt]);
    }
    __syncthreads();   // protect buffer before it is restaged
  }

  // ---- normalize by 1/l and store merged-head bf16 [B,S,D] ----
  aw[row] = 1.0f / lrun;
  wait_ds0();
  const int b = bh >> 4, h = bh & 15;
#pragma unroll
  for (int r = 0; r < 8; ++r) {
    const float rn = aw[half * 8 + r];
    const int srow = qbase + half * 8 + r;
    const size_t ob = ((size_t)(b * HS + srow)) * HD + h * HDK;
#pragma unroll
    for (int dt = 0; dt < 4; ++dt)
      O[ob + dt * 16 + col] = (__bf16)(o[dt][r] * rn);
  }
}

// ---------------- host-side orchestration ----------------

extern "C" void kernel_launch(void* const* d_in, const int* in_sizes, int n_in,
                              void* d_out, int out_size, void* d_ws, size_t ws_size,
                              hipStream_t stream) {
  const float* q  = (const float*)d_in[0];
  const float* k  = (const float*)d_in[1];
  const float* v  = (const float*)d_in[2];
  const float* Wq = (const float*)d_in[3];
  const float* Wk = (const float*)d_in[4];
  const float* Wv = (const float*)d_in[5];
  const float* Wo = (const float*)d_in[6];
  const float* qw = (const float*)d_in[7];
  const float* kw = (const float*)d_in[8];

  const size_t nMD = (size_t)HM * HD;   // 4096*1024
  const size_t nDD = (size_t)HD * HD;   // 1024*1024

  char* ws = (char*)d_ws;
  __bf16* Xbf = (__bf16*)ws; ws += 3 * nMD * sizeof(__bf16);   // q,k,v bf16
  __bf16* Wbf = (__bf16*)ws; ws += 4 * nDD * sizeof(__bf16);   // Wq,Wk,Wv,Wo bf16
  float*  Pf  = (float*)ws;  ws += 3 * nMD * sizeof(float);    // projection outputs f32
  __bf16* Hbf = (__bf16*)ws; ws += 3 * nMD * sizeof(__bf16);   // [B,H,S,DK] q,k,v
  __bf16* Abf = (__bf16*)ws; ws += nMD * sizeof(__bf16);       // merged attn out bf16

  // 0) convert inputs & weights to bf16
  {
    const int n4 = (int)(nMD / 4);
    dim3 g((n4 + 255) / 256);
    cvt_f32_bf16<<<g, 256, 0, stream>>>(q, Xbf + 0 * nMD, n4);
    cvt_f32_bf16<<<g, 256, 0, stream>>>(k, Xbf + 1 * nMD, n4);
    cvt_f32_bf16<<<g, 256, 0, stream>>>(v, Xbf + 2 * nMD, n4);
    const int w4 = (int)(nDD / 4);
    dim3 gw((w4 + 255) / 256);
    cvt_f32_bf16<<<gw, 256, 0, stream>>>(Wq, Wbf + 0 * nDD, w4);
    cvt_f32_bf16<<<gw, 256, 0, stream>>>(Wk, Wbf + 1 * nDD, w4);
    cvt_f32_bf16<<<gw, 256, 0, stream>>>(Wv, Wbf + 2 * nDD, w4);
    cvt_f32_bf16<<<gw, 256, 0, stream>>>(Wo, Wbf + 3 * nDD, w4);
  }

  // 1) Q/K/V projections (WMMA, double-buffered async-LDS staging)
  dim3 gg(HD / 128, HM / 128);
  for (int i = 0; i < 3; ++i)
    gemm_bf16_nt<<<gg, 256, 0, stream>>>(Xbf + i * nMD, Wbf + i * nDD, Pf + i * nMD,
                                         HM, HD, HD);

  // 2) per-head RMSNorm + RoPE (+relayout); V gets plain relayout
  dim3 gn((HM * HH) / 8);
  head_norm_rope<<<gn, 256, 0, stream>>>(Pf + 0 * nMD, qw, Hbf + 0 * nMD, 1);
  head_norm_rope<<<gn, 256, 0, stream>>>(Pf + 1 * nMD, kw, Hbf + 1 * nMD, 1);
  head_norm_rope<<<gn, 256, 0, stream>>>(Pf + 2 * nMD, nullptr, Hbf + 2 * nMD, 0);

  // 3) causal flash attention (WMMA)
  dim3 ga(HS / 64, HB * HH);
  flash_attn<<<ga, 128, 0, stream>>>(Hbf + 0 * nMD, Hbf + 1 * nMD, Hbf + 2 * nMD, Abf);

  // 4) output projection (WMMA) -> f32 d_out
  gemm_bf16_nt<<<gg, 256, 0, stream>>>(Abf, Wbf + 3 * nDD, (float*)d_out, HM, HD, HD);

  (void)in_sizes; (void)n_in; (void)out_size; (void)ws_size;
}
